// MonotonicAlignmentSearch_35296041239267
// MI455X (gfx1250) — matrix operand order repaired and checked by reading, not compiled, not run on hardware
//
#include <hip/hip_runtime.h>
#include <hip/hip_bf16.h>

typedef __attribute__((ext_vector_type(16))) __bf16 v16bf;
typedef __attribute__((ext_vector_type(8)))  __bf16 v8bf;
typedef __attribute__((ext_vector_type(8)))  float  v8f;

#define HH 1024
#define TTT 128
#define TAA 512
#define NB 2
#define NT 4                       // N-tiles (of 16) per wave
#define KF (HH / 32)               // 32 K-chunks
#define NGRP ((HH / 16) / NT)      // 16 n-groups
#define PLANE ((long)(HH / 16) * KF * 512)  // bf16 elems per packed [K=1024,N=1024] plane

static __device__ inline v16bf cat8(v8bf a, v8bf b) {
  v16bf o;
#pragma unroll
  for (int i = 0; i < 8; ++i) { o[i] = a[i]; o[8 + i] = b[i]; }
  return o;
}

// A fragment: 16x32 bf16 from row-major bf16 (rows pre-validated / zero-padded).
// Per ISA layout, lane (half,m) needs K = half*8+{0..7} and 16+half*8+{0..7}:
// two contiguous 16-byte segments.
static __device__ inline v16bf load_a_bf(const __bf16* __restrict__ A, int lda) {
  int lane = threadIdx.x & 31;
  int half = lane >> 4;
  int m    = lane & 15;
  const __bf16* p = A + (long)m * lda + half * 8;
  return cat8(*(const v8bf*)p, *(const v8bf*)(p + 16));
}

// B fragment from pre-packed buffer (fragment-major, 512 bf16 per fragment).
static __device__ inline v16bf load_b_packed(const __bf16* __restrict__ P, long frag) {
  int lane = threadIdx.x & 31;
  const __bf16* p = P + (frag * 32 + lane) * 16;
  return cat8(*(const v8bf*)p, *(const v8bf*)(p + 8));
}

// ---- Pack fp32 weight (element (k,n) at W[k*sK + n*sN]) into WMMA B-fragment order. ----
__global__ void pack_b_kernel(const float* __restrict__ W, long sK, long sN,
                              __bf16* __restrict__ out) {
  int wave = threadIdx.x >> 5;
  long f = (long)blockIdx.x * (blockDim.x >> 5) + wave;   // f = nt*KF + kc
  long nfrag = (long)(HH / 16) * KF;
  if (f >= nfrag) return;
  int kc = (int)(f % KF), nt = (int)(f / KF);
  int lane  = threadIdx.x & 31;
  int n     = nt * 16 + (lane & 15);
  int khalf = lane >> 4;
  __bf16* o = out + (f * 32 + lane) * 16;
#pragma unroll
  for (int v = 0; v < 8; ++v) {
    int k = kc * 32 + khalf * 8 + 2 * v + (v >= 4 ? 8 : 0);
    o[2 * v]     = (__bf16)W[(long)k * sK + (long)n * sN];
    o[2 * v + 1] = (__bf16)W[(long)(k + 1) * sK + (long)n * sN];
  }
}

// ---- Convert fp32 [B,rows,H] -> bf16 [B,rows+2*pad,H] with zero pad rows. ----
__global__ void convert_pad_kernel(const float* __restrict__ X, __bf16* __restrict__ out,
                                   int rows, int pad) {
  int prows = rows + 2 * pad;
  long total = (long)NB * prows * HH;
  for (long i = (long)blockIdx.x * blockDim.x + threadIdx.x; i < total;
       i += (long)gridDim.x * blockDim.x) {
    int  c  = (int)(i % HH);
    long rt = i / HH;
    int  t  = (int)(rt % prows);
    int  b  = (int)(rt / prows);
    int  ts = t - pad;
    float v = 0.f;
    if (ts >= 0 && ts < rows) v = X[((long)b * rows + ts) * HH + c];
    out[i] = (__bf16)v;
  }
}

// ---- Apply GroupNorm(1) and write padded bf16 [B,TT+2,H]. ----
__global__ void gn_apply_pad_kernel(const float* __restrict__ Y, const float* __restrict__ stats,
                                    const float* __restrict__ g, const float* __restrict__ bt,
                                    __bf16* __restrict__ out) {
  long total = (long)NB * (TTT + 2) * HH;
  for (long i = (long)blockIdx.x * blockDim.x + threadIdx.x; i < total;
       i += (long)gridDim.x * blockDim.x) {
    int  c  = (int)(i % HH);
    long rt = i / HH;
    int  t  = (int)(rt % (TTT + 2));
    int  b  = (int)(rt / (TTT + 2));
    float v = 0.f;
    if (t >= 1 && t <= TTT) {
      float mu = stats[2 * b], rs = stats[2 * b + 1];
      float x  = Y[((long)b * TTT + (t - 1)) * HH + c];
      v = (x - mu) * rs * g[c] + bt[c];
    }
    out[i] = (__bf16)v;
  }
}

// ---- GEMM: C[b][m][n] = A[b][m][:] @ Bpacked (+bias). K=N=1024. One wave: 16x64 C. ----
__global__ void gemm_bf_kernel(const __bf16* __restrict__ A, long a_bstride, int mtiles, int batches,
                               const __bf16* __restrict__ Bp, const float* __restrict__ bias,
                               float* __restrict__ C, long c_bstride) {
  int wave = threadIdx.x >> 5;
  int task = blockIdx.x * (blockDim.x >> 5) + wave;
  int ng   = task % NGRP;
  int rest = task / NGRP;
  int mt   = rest % mtiles;
  int b    = rest / mtiles;
  if (b >= batches) return;
  const __bf16* Ab = A + (long)b * a_bstride + (long)mt * 16 * HH;
  v8f acc[NT] = {};
  for (int kc = 0; kc < KF; ++kc) {
    v16bf a = load_a_bf(Ab + kc * 32, HH);
#pragma unroll
    for (int j = 0; j < NT; ++j) {
      v16bf bf = load_b_packed(Bp, (long)(ng * NT + j) * KF + kc);
      acc[j] = __builtin_amdgcn_wmma_f32_16x16x32_bf16(false, a, false, bf, (short)0, acc[j], false, false);
    }
  }
  int lane = threadIdx.x & 31;
#pragma unroll
  for (int j = 0; j < NT; ++j) {
    int n = (ng * NT + j) * 16 + (lane & 15);
    float bv = bias ? bias[n] : 0.f;
#pragma unroll
    for (int v = 0; v < 8; ++v) {
      int m = mt * 16 + v + 8 * (lane >> 4);
      C[(long)b * c_bstride + (long)m * HH + n] = acc[j][v] + bv;
    }
  }
}

// ---- conv1d(K=3) as 3 shifted GEMMs over padded bf16 input, + bias + ReLU. ----
__global__ void conv_gemm_bf_kernel(const __bf16* __restrict__ Xpad,  // [B,TT+2,H]
                                    const __bf16* __restrict__ Wp,    // 3 packed planes
                                    const float* __restrict__ bias,
                                    float* __restrict__ Y) {          // [B,TT,H]
  int wave = threadIdx.x >> 5;
  int task = blockIdx.x * (blockDim.x >> 5) + wave;
  int ng   = task % NGRP;
  int rest = task / NGRP;
  int tt   = rest % (TTT / 16);
  int b    = rest / (TTT / 16);
  if (b >= NB) return;
  v8f acc[NT] = {};
  for (int kk = 0; kk < 3; ++kk) {
    const __bf16* Ab = Xpad + ((long)b * (TTT + 2) + tt * 16 + kk) * HH;  // +1 pad, -1 shift
    const __bf16* Wk = Wp + (long)kk * PLANE;
    for (int kc = 0; kc < KF; ++kc) {
      v16bf a = load_a_bf(Ab + kc * 32, HH);
#pragma unroll
      for (int j = 0; j < NT; ++j) {
        v16bf bf = load_b_packed(Wk, (long)(ng * NT + j) * KF + kc);
        acc[j] = __builtin_amdgcn_wmma_f32_16x16x32_bf16(false, a, false, bf, (short)0, acc[j], false, false);
      }
    }
  }
  int lane = threadIdx.x & 31;
#pragma unroll
  for (int j = 0; j < NT; ++j) {
    int o = (ng * NT + j) * 16 + (lane & 15);
    float bv = bias[o];
#pragma unroll
    for (int v = 0; v < 8; ++v) {
      int t = tt * 16 + v + 8 * (lane >> 4);
      Y[((long)b * TTT + t) * HH + o] = fmaxf(acc[j][v] + bv, 0.f);
    }
  }
}

// ---- Per-batch GroupNorm(1) stats: mean + rsqrt(var+eps). ----
__global__ void gn_stats_kernel(const float* __restrict__ Y, float* __restrict__ stats) {
  int b = blockIdx.x;
  const float* p = Y + (long)b * TTT * HH;
  float s = 0.f, sq = 0.f;
  for (int i = threadIdx.x; i < TTT * HH; i += 256) { float v = p[i]; s += v; sq += v * v; }
  __shared__ float rs[256], rq[256];
  rs[threadIdx.x] = s; rq[threadIdx.x] = sq;
  __syncthreads();
  for (int off = 128; off > 0; off >>= 1) {
    if (threadIdx.x < off) { rs[threadIdx.x] += rs[threadIdx.x + off]; rq[threadIdx.x] += rq[threadIdx.x + off]; }
    __syncthreads();
  }
  if (threadIdx.x == 0) {
    float n   = (float)(TTT * HH);
    float mu  = rs[0] / n;
    float var = rq[0] / n - mu * mu;
    stats[2 * b]     = mu;
    stats[2 * b + 1] = rsqrtf(var + 1e-5f);
  }
}

// ---- Final 1x1 conv (H->1) on GN'd input + softplus -> durations. ----
__global__ void duration_kernel(const float* __restrict__ Y2, const float* __restrict__ stats2,
                                const float* __restrict__ g2, const float* __restrict__ bt2,
                                const float* __restrict__ w3, const float* __restrict__ b3,
                                float* __restrict__ dur) {
  int row = blockIdx.x;  // b*TT + t
  int b   = row / TTT;
  float mu = stats2[2 * b], rsg = stats2[2 * b + 1];
  const float* p = Y2 + (long)row * HH;
  float s = 0.f;
  for (int c = threadIdx.x; c < HH; c += 256) {
    float xn = (p[c] - mu) * rsg * g2[c] + bt2[c];
    s = fmaf(xn, w3[c], s);
  }
  __shared__ float red[256];
  red[threadIdx.x] = s; __syncthreads();
  for (int off = 128; off > 0; off >>= 1) {
    if (threadIdx.x < off) red[threadIdx.x] += red[threadIdx.x + off];
    __syncthreads();
  }
  if (threadIdx.x == 0) {
    float x = red[0] + b3[0];
    dur[row] = (x > 20.f) ? x : log1pf(expf(x));  // softplus
  }
}

// ---- Fused relu-bilinear logits (no [B,TT,TA,H] materialization). ----
#define DC 64
__global__ void bilinear_logits_kernel(const float* __restrict__ tp,  // [B,TT,H] incl a_b1
                                       const float* __restrict__ ap,  // [B,TA,H]
                                       const float* __restrict__ w2,  // [H]
                                       const float* __restrict__ b2,  // [1]
                                       float* __restrict__ logits) {  // [B,TT,TA]
  int at   = blockIdx.x % (TAA / 16);
  int rest = blockIdx.x / (TAA / 16);
  int tt   = rest % (TTT / 16);
  int b    = rest / (TTT / 16);
  __shared__ float s_tp[16][DC + 1];
  __shared__ float s_ap[16][DC + 1];
  __shared__ float s_w2[DC];
  int tid = threadIdx.x;
  int ti  = tid >> 4;
  int ai  = tid & 15;
  float acc = 0.f;
  for (int d0 = 0; d0 < HH; d0 += DC) {
    __syncthreads();
    for (int idx = tid; idx < 16 * DC; idx += 256) {
      int r = idx / DC, c = idx % DC;
      s_tp[r][c] = tp[((long)b * TTT + tt * 16 + r) * HH + d0 + c];
      s_ap[r][c] = ap[((long)b * TAA + at * 16 + r) * HH + d0 + c];
    }
    if (tid < DC) s_w2[tid] = w2[d0 + tid];
    __syncthreads();
#pragma unroll 8
    for (int c = 0; c < DC; ++c)
      acc = fmaf(fmaxf(s_tp[ti][c] + s_ap[ai][c], 0.f), s_w2[c], acc);
  }
  int t = tt * 16 + ti;
  int a = at * 16 + ai;
  float mono = -0.1f * fabsf((float)a - (float)t * ((float)TAA / (float)TTT));
  logits[((long)b * TTT + t) * TAA + a] = acc + b2[0] + mono;
}

// ---- Row softmax over TA, in place on d_out. ----
__global__ void softmax_kernel(float* __restrict__ logits) {
  int row = blockIdx.x;
  float* p = logits + (long)row * TAA;
  __shared__ float red[256];
  int tid = threadIdx.x;
  float m = -1e30f;
  for (int i = tid; i < TAA; i += 256) m = fmaxf(m, p[i]);
  red[tid] = m; __syncthreads();
  for (int off = 128; off > 0; off >>= 1) { if (tid < off) red[tid] = fmaxf(red[tid], red[tid + off]); __syncthreads(); }
  m = red[0]; __syncthreads();
  float vals[TAA / 256];
  float s = 0.f;
  for (int i = tid, j = 0; i < TAA; i += 256, ++j) { float e = expf(p[i] - m); vals[j] = e; s += e; }
  red[tid] = s; __syncthreads();
  for (int off = 128; off > 0; off >>= 1) { if (tid < off) red[tid] += red[tid + off]; __syncthreads(); }
  float inv = 1.f / red[0];
  for (int i = tid, j = 0; i < TAA; i += 256, ++j) p[i] = vals[j] * inv;
}

extern "C" void kernel_launch(void* const* d_in, const int* in_sizes, int n_in,
                              void* d_out, int out_size, void* d_ws, size_t ws_size,
                              hipStream_t stream) {
  (void)in_sizes; (void)n_in; (void)out_size; (void)ws_size;
  const float* text  = (const float*)d_in[0];
  const float* audio = (const float*)d_in[1];
  const float* a_w1  = (const float*)d_in[2];
  const float* a_b1  = (const float*)d_in[3];
  const float* a_w2  = (const float*)d_in[4];
  const float* a_b2  = (const float*)d_in[5];
  const float* d_w1  = (const float*)d_in[6];
  const float* d_b1  = (const float*)d_in[7];
  const float* gn1_g = (const float*)d_in[8];
  const float* gn1_b = (const float*)d_in[9];
  const float* d_w2  = (const float*)d_in[10];
  const float* d_b2  = (const float*)d_in[11];
  const float* gn2_g = (const float*)d_in[12];
  const float* gn2_b = (const float*)d_in[13];
  const float* d_w3  = (const float*)d_in[14];
  const float* d_b3  = (const float*)d_in[15];

  float* out = (float*)d_out;
  float* alignment = out;                        // [2,128,512]
  float* durations = out + (long)NB * TTT * TAA; // [2,128]

  // fp32 workspace
  float* ws  = (float*)d_ws;
  float* tp  = ws;                          // 2*128*1024
  float* ap  = tp + (long)NB * TTT * HH;    // 2*512*1024
  float* y1  = ap + (long)NB * TAA * HH;    // 2*128*1024
  float* y2  = y1 + (long)NB * TTT * HH;    // 2*128*1024
  float* st1 = y2 + (long)NB * TTT * HH;    // 4
  float* st2 = st1 + 4;                     // 4
  // bf16 workspace (after fp32 region)
  __bf16* bfw      = (__bf16*)(st2 + 4);
  __bf16* text_pad = bfw;                                   // 2*130*1024
  __bf16* audio_bf = text_pad + (long)NB * (TTT + 2) * HH;  // 2*512*1024
  __bf16* y1n_pad  = audio_bf + (long)NB * TAA * HH;        // 2*130*1024
  __bf16* Wa_t     = y1n_pad + (long)NB * (TTT + 2) * HH;   // 1 plane
  __bf16* Wa_a     = Wa_t + PLANE;                          // 1 plane
  __bf16* Wc1      = Wa_a + PLANE;                          // 3 planes
  __bf16* Wc2      = Wc1 + 3 * PLANE;                       // 3 planes

  dim3 blk(256);

  // ---- preprocessing: convert/pad activations, pack weights ----
  convert_pad_kernel<<<1040, blk, 0, stream>>>(text, text_pad, TTT, 1);
  convert_pad_kernel<<<4096, blk, 0, stream>>>(audio, audio_bf, TAA, 0);
  pack_b_kernel<<<256, blk, 0, stream>>>(a_w1, HH, 1, Wa_t);
  pack_b_kernel<<<256, blk, 0, stream>>>(a_w1 + (long)HH * HH, HH, 1, Wa_a);
  for (int kk = 0; kk < 3; ++kk) {
    pack_b_kernel<<<256, blk, 0, stream>>>(d_w1 + kk, 3, 3 * HH, Wc1 + (long)kk * PLANE);
    pack_b_kernel<<<256, blk, 0, stream>>>(d_w2 + kk, 3, 3 * HH, Wc2 + (long)kk * PLANE);
  }

  // ---- alignment projections ----
  // tp = text @ a_w1[:H] + a_b1  (per-batch M=128; skip pad row with +HH)
  gemm_bf_kernel<<<32, blk, 0, stream>>>(text_pad + HH, (long)(TTT + 2) * HH, TTT / 16, NB,
                                         Wa_t, a_b1, tp, (long)TTT * HH);
  // ap = audio @ a_w1[H:]        (single batch M=1024)
  gemm_bf_kernel<<<128, blk, 0, stream>>>(audio_bf, 0, (NB * TAA) / 16, 1,
                                          Wa_a, nullptr, ap, 0);

  // ---- duration branch ----
  conv_gemm_bf_kernel<<<32, blk, 0, stream>>>(text_pad, Wc1, d_b1, y1);
  gn_stats_kernel<<<NB, blk, 0, stream>>>(y1, st1);
  gn_apply_pad_kernel<<<1040, blk, 0, stream>>>(y1, st1, gn1_g, gn1_b, y1n_pad);
  conv_gemm_bf_kernel<<<32, blk, 0, stream>>>(y1n_pad, Wc2, d_b2, y2);
  gn_stats_kernel<<<NB, blk, 0, stream>>>(y2, st2);
  duration_kernel<<<NB * TTT, blk, 0, stream>>>(y2, st2, gn2_g, gn2_b, d_w3, d_b3, durations);

  // ---- alignment branch ----
  bilinear_logits_kernel<<<NB * (TTT / 16) * (TAA / 16), blk, 0, stream>>>(tp, ap, a_w2, a_b2, alignment);
  softmax_kernel<<<NB * TTT, blk, 0, stream>>>(alignment);
}
